// FlowV2Attention_84593675862629
// MI455X (gfx1250) — matrix operand order repaired and checked by reading, not compiled, not run on hardware
//
#include <hip/hip_runtime.h>
#include <hip/hip_bf16.h>

#define DIMV   1024
#define NHEADS 16
#define HDIM   64
#define BATCH  4
#define SEQ    2048

typedef __bf16        bf16x16 __attribute__((ext_vector_type(16)));
typedef float         f32x8   __attribute__((ext_vector_type(8)));
typedef unsigned int  u32x4   __attribute__((ext_vector_type(4)));

union FragU { u32x4 u[2]; bf16x16 v; };

__device__ __forceinline__ unsigned short f2bf(float f) {
  unsigned int u = __float_as_uint(f);
  u += 0x7FFFu + ((u >> 16) & 1u);   // round-to-nearest-even
  return (unsigned short)(u >> 16);
}
__device__ __forceinline__ float bf2f(unsigned short h) {
  return __uint_as_float(((unsigned int)h) << 16);
}

// A-matrix 16x32 bf16 fragment (row per lane l%16), chunks at K = 8*(l/16)+e and 16+8*(l/16)+e
__device__ __forceinline__ bf16x16 ldfragA(const unsigned short* rowp, int half) {
  const unsigned short* p = rowp + half * 8;
  FragU f;
  f.u[0] = *(const u32x4*)(p);
  f.u[1] = *(const u32x4*)(p + 16);
  return f.v;
}
// B-matrix 32x16 bf16 fragment (col per lane l%16), chunks at K = 8*(l/16)+e and 16+8*(l/16)+e
__device__ __forceinline__ bf16x16 ldfragB(const unsigned short* rowp, int half) {
  const unsigned short* p = rowp + half * 8;
  FragU f;
  f.u[0] = *(const u32x4*)(p);
  f.u[1] = *(const u32x4*)(p + 16);
  return f.v;
}

#define WMMA_BF16(a, b, c) \
  __builtin_amdgcn_wmma_f32_16x16x32_bf16(false, (a), false, (b), (short)0, (c), false, false)

// ---------------- conversion kernels ----------------
__global__ __launch_bounds__(256) void cvt_bf16_kernel(
    const float* __restrict__ in, unsigned short* __restrict__ out, size_t n) {
  size_t i = (size_t)blockIdx.x * blockDim.x + threadIdx.x;
  size_t stride = (size_t)gridDim.x * blockDim.x;
  for (; i < n; i += stride) out[i] = f2bf(in[i]);
}

__global__ __launch_bounds__(256) void cvt_transpose_kernel(
    const float* __restrict__ W, unsigned short* __restrict__ Wt, int K, int N) {
  size_t i = (size_t)blockIdx.x * blockDim.x + threadIdx.x;
  size_t total = (size_t)K * N;
  size_t stride = (size_t)gridDim.x * blockDim.x;
  for (; i < total; i += stride) {
    size_t k = i / (size_t)N, n = i % (size_t)N;
    Wt[n * (size_t)K + k] = f2bf(W[i]);
  }
}

// ---------------- WMMA GEMM: C[M,N] = A_bf[M,K] @ Bt_bf[N,K]^T + bias ----------------
// Block tile 128(M) x 256(N), 8 waves in 2x4 grid, 64x64 per wave (16 WMMA / k-step).
__global__ __launch_bounds__(256) void wmma_gemm_kernel(
    const unsigned short* __restrict__ A,
    const unsigned short* __restrict__ Bt,
    const float* __restrict__ bias,
    float* __restrict__ Cf,
    unsigned short* __restrict__ Cbf,
    int M, int N, int K, int store_bf16) {
  const int lane = threadIdx.x & 31;
  const int w    = threadIdx.x >> 5;
  const int half = lane >> 4;
  const int ln   = lane & 15;
  const int m0 = blockIdx.y * 128 + (w >> 2) * 64;
  const int n0 = blockIdx.x * 256 + (w & 3) * 64;

  const unsigned short* Arow[4];
  const unsigned short* Brow[4];
#pragma unroll
  for (int i = 0; i < 4; ++i) {
    Arow[i] = A  + (size_t)(m0 + i * 16 + ln) * K;
    Brow[i] = Bt + (size_t)(n0 + i * 16 + ln) * K;
  }

  f32x8 c[4][4] = {};
  for (int k0 = 0; k0 < K; k0 += 32) {
    __builtin_prefetch(Arow[0] + k0 + 256, 0, 1);
    __builtin_prefetch(Brow[0] + k0 + 256, 0, 1);
    bf16x16 a[4], b[4];
#pragma unroll
    for (int i = 0; i < 4; ++i) a[i] = ldfragA(Arow[i] + k0, half);
#pragma unroll
    for (int j = 0; j < 4; ++j) b[j] = ldfragB(Brow[j] + k0, half);
#pragma unroll
    for (int i = 0; i < 4; ++i)
#pragma unroll
      for (int j = 0; j < 4; ++j)
        c[i][j] = WMMA_BF16(a[i], b[j], c[i][j]);
  }

  float bv[4];
#pragma unroll
  for (int j = 0; j < 4; ++j) bv[j] = bias ? bias[n0 + j * 16 + ln] : 0.f;

#pragma unroll
  for (int i = 0; i < 4; ++i) {
#pragma unroll
    for (int r = 0; r < 8; ++r) {
      int m = m0 + i * 16 + r + 8 * half;
#pragma unroll
      for (int j = 0; j < 4; ++j) {
        float v = c[i][j][r] + bv[j];
        size_t idx = (size_t)m * N + n0 + j * 16 + ln;
        if (store_bf16) Cbf[idx] = f2bf(v);
        else            Cf[idx]  = v;
      }
    }
  }
}

// ---------------- RoPE + rearrange ----------------
// qkv bf16 [B*T, 3*D] -> Qr,Kr bf16 [B,H,T,hd] (rotated), Vt bf16 [B,H,hd,T]
__global__ __launch_bounds__(256) void rope_rearrange_kernel(
    const unsigned short* __restrict__ qkv,
    unsigned short* __restrict__ Qr,
    unsigned short* __restrict__ Kr,
    unsigned short* __restrict__ Vt) {
  size_t tid = (size_t)blockIdx.x * blockDim.x + threadIdx.x;
  int i = (int)(tid & 31);          // rotation pair index 0..31
  int h = (int)((tid >> 5) & 15);
  int t = (int)((tid >> 9) & 2047);
  int b = (int)(tid >> 20);
  if (b >= BATCH) return;

  const float LOG1E4 = 9.210340371976184f;  // ln(10000)
  float inv_freq = __expf(-LOG1E4 * (float)(2 * i) / (float)HDIM);
  float ang = (float)t * inv_freq;
  float c = __cosf(ang), s = __sinf(ang);

  size_t m = (size_t)b * SEQ + t;
  const unsigned short* row = qkv + m * (3 * DIMV);
  int col = h * HDIM + 2 * i;
  float q0 = bf2f(row[col]),            q1 = bf2f(row[col + 1]);
  float k0 = bf2f(row[DIMV + col]),     k1 = bf2f(row[DIMV + col + 1]);
  float v0 = bf2f(row[2 * DIMV + col]), v1 = bf2f(row[2 * DIMV + col + 1]);

  float qo0 = q0 * c - q1 * s, qo1 = q0 * s + q1 * c;
  float ko0 = k0 * c - k1 * s, ko1 = k0 * s + k1 * c;

  size_t bh = (size_t)b * NHEADS + h;
  size_t qi = (bh * SEQ + t) * HDIM + 2 * i;
  Qr[qi] = f2bf(qo0); Qr[qi + 1] = f2bf(qo1);
  Kr[qi] = f2bf(ko0); Kr[qi + 1] = f2bf(ko1);
  size_t vbase = bh * (size_t)HDIM * SEQ;
  Vt[vbase + (size_t)(2 * i) * SEQ + t]     = f2bf(v0);
  Vt[vbase + (size_t)(2 * i + 1) * SEQ + t] = f2bf(v1);
}

// ---------------- flash attention: one wave per 16-query tile, 64 keys/step ----------------
__global__ __launch_bounds__(256) void attn_kernel(
    const unsigned short* __restrict__ Qr,
    const unsigned short* __restrict__ Kr,
    const unsigned short* __restrict__ Vt,
    unsigned short* __restrict__ Obf) {
  __shared__ __align__(16) unsigned short pbuf[8][16 * 64];  // per-wave P patch (2KB)

  const int lane = threadIdx.x & 31;
  const int w    = threadIdx.x >> 5;
  const int half = lane >> 4;
  const int ln   = lane & 15;
  const int nq   = SEQ / 16;
  int g  = blockIdx.x * 8 + w;
  int qt = g % nq;
  int bh = g / nq;
  if (bh >= BATCH * NHEADS) return;

  const unsigned short* Qp = Qr + (size_t)bh * SEQ * HDIM;
  const unsigned short* Kp = Kr + (size_t)bh * SEQ * HDIM;
  const unsigned short* Vp = Vt + (size_t)bh * HDIM * SEQ;

  const unsigned short* qrow = Qp + (size_t)(qt * 16 + ln) * HDIM;
  bf16x16 qf0 = ldfragA(qrow, half);        // hd 0..31
  bf16x16 qf1 = ldfragA(qrow + 32, half);   // hd 32..63

  f32x8 o[4] = {};
  float mrun[8], lrun[8];
#pragma unroll
  for (int r = 0; r < 8; ++r) { mrun[r] = -1e30f; lrun[r] = 0.f; }

  unsigned short* pb = &pbuf[w][0];

  for (int kt = 0; kt < SEQ; kt += 64) {
    // four 16x16 score tiles (keys kt .. kt+63)
    f32x8 s[4];
#pragma unroll
    for (int ct = 0; ct < 4; ++ct) {
      const unsigned short* krow = Kp + (size_t)(kt + ct * 16 + ln) * HDIM;
      f32x8 z = {};
      s[ct] = WMMA_BF16(qf0, ldfragB(krow, half), z);
      s[ct] = WMMA_BF16(qf1, ldfragB(krow + 32, half), s[ct]);
      s[ct] = s[ct] * 0.125f;   // 1/sqrt(64)
    }

#pragma unroll
    for (int r = 0; r < 8; ++r) {
      float t0 = fmaxf(fmaxf(s[0][r], s[1][r]), fmaxf(s[2][r], s[3][r]));
      t0 = fmaxf(t0, __shfl_xor(t0, 8));
      t0 = fmaxf(t0, __shfl_xor(t0, 4));
      t0 = fmaxf(t0, __shfl_xor(t0, 2));
      t0 = fmaxf(t0, __shfl_xor(t0, 1));
      float mn = fmaxf(mrun[r], t0);
      float al = __expf(mrun[r] - mn);
      mrun[r] = mn;
      float p0 = __expf(s[0][r] - mn);
      float p1 = __expf(s[1][r] - mn);
      float p2 = __expf(s[2][r] - mn);
      float p3 = __expf(s[3][r] - mn);
      float rs = (p0 + p1) + (p2 + p3);
      rs += __shfl_xor(rs, 8);
      rs += __shfl_xor(rs, 4);
      rs += __shfl_xor(rs, 2);
      rs += __shfl_xor(rs, 1);
      lrun[r] = lrun[r] * al + rs;
      o[0][r] *= al; o[1][r] *= al; o[2][r] *= al; o[3][r] *= al;
      int row = r + 8 * half;                     // C-layout row for this lane
      pb[row * 64 + ln]      = f2bf(p0);
      pb[row * 64 + 16 + ln] = f2bf(p1);
      pb[row * 64 + 32 + ln] = f2bf(p2);
      pb[row * 64 + 48 + ln] = f2bf(p3);
    }
    asm volatile("s_wait_dscnt 0" ::: "memory");  // LDS store->load, same wave

    // reload P transposed into two A-fragments (keys kt..+31, kt+32..+63)
    FragU pf0, pf1;
    const unsigned short* pr = pb + ln * 64 + half * 8;
    pf0.u[0] = *(const u32x4*)(pr);
    pf0.u[1] = *(const u32x4*)(pr + 16);
    pf1.u[0] = *(const u32x4*)(pr + 32);
    pf1.u[1] = *(const u32x4*)(pr + 48);

#pragma unroll
    for (int j = 0; j < 4; ++j) {
      const unsigned short* vrow = Vp + (size_t)(j * 16 + ln) * SEQ + kt;
      o[j] = WMMA_BF16(pf0.v, ldfragB(vrow, half), o[j]);
      o[j] = WMMA_BF16(pf1.v, ldfragB(vrow + 32, half), o[j]);
    }
  }

  int b = bh / NHEADS, h = bh % NHEADS;
#pragma unroll
  for (int r = 0; r < 8; ++r) {
    float inv = 1.0f / lrun[r];
    int t = qt * 16 + r + 8 * half;
    size_t base = ((size_t)b * SEQ + t) * DIMV + h * HDIM;
    Obf[base + ln]      = f2bf(o[0][r] * inv);
    Obf[base + 16 + ln] = f2bf(o[1][r] * inv);
    Obf[base + 32 + ln] = f2bf(o[2][r] * inv);
    Obf[base + 48 + ln] = f2bf(o[3][r] * inv);
  }
}

// ---------------- host-side orchestration ----------------
extern "C" void kernel_launch(void* const* d_in, const int* in_sizes, int n_in,
                              void* d_out, int out_size, void* d_ws, size_t ws_size,
                              hipStream_t stream) {
  const float* x    = (const float*)d_in[0];
  const float* Wqkv = (const float*)d_in[1];
  const float* bqkv = (const float*)d_in[2];
  const float* Wout = (const float*)d_in[3];
  const float* bout = (const float*)d_in[4];
  float* out = (float*)d_out;

  const size_t M = (size_t)BATCH * SEQ;  // 8192
  char* ws = (char*)d_ws;
  size_t off = 0;
  unsigned short* xb    = (unsigned short*)(ws + off); off += M * DIMV * 2;                // 16 MB
  unsigned short* wqkvT = (unsigned short*)(ws + off); off += (size_t)3 * DIMV * DIMV * 2; // 6 MB
  unsigned short* woutT = (unsigned short*)(ws + off); off += (size_t)DIMV * DIMV * 2;     // 2 MB
  unsigned short* qkvb  = (unsigned short*)(ws + off); off += M * 3 * DIMV * 2;            // 48 MB
  unsigned short* Qrb   = (unsigned short*)(ws + off); off += M * DIMV * 2;                // 16 MB
  unsigned short* Krb   = (unsigned short*)(ws + off); off += M * DIMV * 2;                // 16 MB
  unsigned short* Vtb   = (unsigned short*)(ws + off); off += M * DIMV * 2;                // 16 MB
  unsigned short* Obf   = (unsigned short*)(ws + off); off += M * DIMV * 2;                // 16 MB
  (void)ws_size; (void)in_sizes; (void)n_in; (void)out_size;

  cvt_bf16_kernel<<<2048, 256, 0, stream>>>(x, xb, M * DIMV);
  cvt_transpose_kernel<<<2048, 256, 0, stream>>>(Wqkv, wqkvT, DIMV, 3 * DIMV);
  cvt_transpose_kernel<<<1024, 256, 0, stream>>>(Wout, woutT, DIMV, DIMV);

  // QKV projection: [8192,1024] x [1024,3072] -> bf16 qkv ; block tile 128x256
  wmma_gemm_kernel<<<dim3(3 * DIMV / 256, M / 128), 256, 0, stream>>>(
      xb, wqkvT, bqkv, nullptr, qkvb, (int)M, 3 * DIMV, DIMV, 1);

  rope_rearrange_kernel<<<(BATCH * SEQ * NHEADS * (HDIM / 2)) / 256, 256, 0, stream>>>(
      qkvb, Qrb, Krb, Vtb);

  // attention: B*H*(T/16) = 8192 waves, 8 waves/block
  attn_kernel<<<(BATCH * NHEADS * (SEQ / 16)) / 8, 256, 0, stream>>>(Qrb, Krb, Vtb, Obf);

  // output projection: [8192,1024] x [1024,1024] -> f32 out ; block tile 128x256
  wmma_gemm_kernel<<<dim3(DIMV / 256, M / 128), 256, 0, stream>>>(
      Obf, woutT, bout, out, nullptr, (int)M, DIMV, DIMV, 0);
}
